// MiniCPMAttention_44083544326588
// MI455X (gfx1250) — matrix operand order repaired and checked by reading, not compiled, not run on hardware
//
#include <hip/hip_runtime.h>
#include <hip/hip_bf16.h>
#include <stdint.h>

// ---------------------------------------------------------------------------
// MiniCPM GQA attention forward for MI455X (gfx1250, wave32, WMMA + TDM).
//   B=2, S=2048, H=2048, NH=32, NKV=8, HD=64, G=4
// d_out = [attn_output (B,S,H) | k (B,NKV,S,HD) | v (B,NKV,S,HD)]  (fp32)
//
// All matmuls use v_wmma_f32_16x16x32_bf16 (f32 accumulate). Attention is a
// flash-style fused kernel (online softmax): the S x S score tensor (2.1 GB)
// never touches HBM. K chunks are staged LDS-side by the Tensor Data Mover
// (tensor_load_to_lds + s_wait_tensorcnt) when the builtin is available.
// ---------------------------------------------------------------------------

typedef __attribute__((ext_vector_type(16))) __bf16        v16bf;
typedef __attribute__((ext_vector_type(2)))  __bf16        v2bf;
typedef __attribute__((ext_vector_type(8)))  float         v8f;
typedef __attribute__((ext_vector_type(8)))  unsigned int  v8u;
typedef __attribute__((ext_vector_type(4)))  unsigned int  u32x4;
typedef __attribute__((ext_vector_type(8)))  int           i32x8;
typedef __attribute__((ext_vector_type(4)))  int           i32x4;

#define HDIM 2048
#define SDIM 2048

// fp32 -> bf16 via native conversion (lets the backend pick v_cvt_*bf16*)
__device__ __forceinline__ unsigned short f2bf(float f) {
  return __builtin_bit_cast(unsigned short, (__bf16)f);
}

// two fp32 -> packed bf16x2
__device__ __forceinline__ unsigned int f2bf2(float lo, float hi) {
#if __has_builtin(__builtin_amdgcn_cvt_pk_bf16_f32)
  return __builtin_bit_cast(unsigned int,
                            __builtin_amdgcn_cvt_pk_bf16_f32(lo, hi));
#else
  v2bf v;
  v[0] = (__bf16)lo;
  v[1] = (__bf16)hi;
  return __builtin_bit_cast(unsigned int, v);
#endif
}

// Build a 16-element bf16 fragment from two runs of 8 consecutive bf16.
__device__ __forceinline__ v16bf make_frag(const unsigned short* p0,
                                           const unsigned short* p1) {
  uint4 a = *reinterpret_cast<const uint4*>(p0);
  uint4 b = *reinterpret_cast<const uint4*>(p1);
  v8u v;
  v[0] = a.x; v[1] = a.y; v[2] = a.z; v[3] = a.w;
  v[4] = b.x; v[5] = b.y; v[6] = b.z; v[7] = b.w;
  return __builtin_bit_cast(v16bf, v);
}

// A-fragment, 16(M) x 32(K) bf16 row-major tile (ISA 7.12.2 layout).
__device__ __forceinline__ v16bf load_afrag(const unsigned short* base, int ld,
                                            int lane) {
  int m  = lane & 15;
  int hf = lane >> 4;
  const unsigned short* r = base + m * ld;
  return make_frag(r + hf * 8, r + 16 + hf * 8);
}

// B-fragment, 32(K) x 16(N) from [n][k] storage; lane = N%16 | half.
__device__ __forceinline__ v16bf load_bfrag(const unsigned short* base, int ld,
                                            int lane) {
  int n  = lane & 15;
  int hf = lane >> 4;
  const unsigned short* r = base + n * ld + hf * 8;
  return make_frag(r, r + 16);
}

// ---------------------------------------------------------------------------
// Kernel A: X[4096,2048] @ W[2048,N] -> bf16 ws [4096,N]; fused RoPE epilogue
// (64-wide N tile == one head, so d and d+32 sit in accumulator tiles t,t+2);
// optional fp32 store in [B, nkv, S, 64] layout (k/v cache outputs).
// ---------------------------------------------------------------------------
__global__ __launch_bounds__(256)
void proj_rope_kernel(const float* __restrict__ X, const float* __restrict__ W,
                      const float* __restrict__ cosT,
                      const float* __restrict__ sinT, int N, int do_rope,
                      unsigned short* __restrict__ out_bf,
                      float* __restrict__ out_f32, int nheads) {
  __shared__ unsigned short As[128 * 32];  // [m][k]
  __shared__ unsigned short Bs[64 * 32];   // [n][k] (transposed tile)

  const int tid  = threadIdx.x;
  const int lane = tid & 31;
  const int wave = tid >> 5;
  const int hf   = lane >> 4;
  const int nlo  = lane & 15;
  const int row0 = blockIdx.x * 128;
  const int head = blockIdx.y;
  const int col0 = head * 64;

  v8f acc[4] = {};

  for (int kb = 0; kb < HDIM / 32; ++kb) {
    // Stage A tile 128x32 (float4 loads -> packed bf16 -> b64 LDS stores)
#pragma unroll
    for (int i = 0; i < 4; ++i) {
      int e  = i * 256 + tid;     // 1024 float4
      int m  = e >> 3, k4 = e & 7;
      float4 x = *reinterpret_cast<const float4*>(
          X + (size_t)(row0 + m) * HDIM + kb * 32 + k4 * 4);
      uint2 p;
      p.x = f2bf2(x.x, x.y);
      p.y = f2bf2(x.z, x.w);
      *reinterpret_cast<uint2*>(&As[m * 32 + k4 * 4]) = p;
    }
    // Stage B tile 32x64, transposed into [n][k]
#pragma unroll
    for (int i = 0; i < 2; ++i) {
      int e = i * 256 + tid;      // 512 float4
      int k = e >> 4, n4 = e & 15;
      float4 wv = *reinterpret_cast<const float4*>(
          &W[(size_t)(kb * 32 + k) * N + col0 + n4 * 4]);
      Bs[(n4 * 4 + 0) * 32 + k] = f2bf(wv.x);
      Bs[(n4 * 4 + 1) * 32 + k] = f2bf(wv.y);
      Bs[(n4 * 4 + 2) * 32 + k] = f2bf(wv.z);
      Bs[(n4 * 4 + 3) * 32 + k] = f2bf(wv.w);
    }
    if (kb + 1 < HDIM / 32) {
      __builtin_prefetch(&X[(size_t)(row0 + (tid >> 1)) * HDIM + (kb + 1) * 32],
                         0, 1);
      __builtin_prefetch(&W[(size_t)((kb + 1) * 32 + (tid & 31)) * N + col0], 0,
                         1);
    }
    __syncthreads();

    v16bf a = load_afrag(&As[wave * 16 * 32], 32, lane);
#pragma unroll
    for (int t = 0; t < 4; ++t) {
      v16bf b = load_bfrag(&Bs[t * 16 * 32], 32, lane);
      acc[t] = __builtin_amdgcn_wmma_f32_16x16x32_bf16(
          false, a, false, b, (short)0, acc[t], false, false);
    }
    __syncthreads();
  }

  // Epilogue: RoPE + stores. C layout: lanes0-15 M=r, lanes16-31 M=r+8.
  const int mbase = row0 + wave * 16;
#pragma unroll
  for (int r = 0; r < 8; ++r) {
    int m   = r + hf * 8;
    int row = mbase + m;
    int s   = row & (SDIM - 1);
    int bb  = row >> 11;  // row / SDIM
    if (do_rope) {
#pragma unroll
      for (int t = 0; t < 2; ++t) {
        int d0   = nlo + 16 * t;
        int d1   = d0 + 32;
        float c0 = cosT[s * 64 + d0], s0 = sinT[s * 64 + d0];
        float c1 = cosT[s * 64 + d1], s1 = sinT[s * 64 + d1];
        float x0 = acc[t][r], x1 = acc[t + 2][r];
        acc[t][r]     = x0 * c0 - x1 * s0;  // q*cos + rot_half(q)*sin, d<32
        acc[t + 2][r] = x1 * c1 + x0 * s1;  // d>=32
      }
    }
#pragma unroll
    for (int t = 0; t < 4; ++t) {
      int d   = nlo + 16 * t;
      float y = acc[t][r];
      out_bf[(size_t)row * N + col0 + d] = f2bf(y);
      if (out_f32) {
        out_f32[(((size_t)bb * nheads + head) * SDIM + s) * 64 + d] = y;
      }
    }
  }
}

// ---------------------------------------------------------------------------
// Kernel B: flash attention, causal, GQA (kv head = q head / 4).
// Grid: (S/128, NH, B); block 256 = 8 waves; wave -> 16 q rows.
// K chunk staged by TDM (tensor_load_to_lds) when available; V staged with a
// manual transpose into [d][key].
// ---------------------------------------------------------------------------
__global__ __launch_bounds__(256)
void flash_attn_kernel(const unsigned short* __restrict__ qws,
                       const unsigned short* __restrict__ kws,
                       const unsigned short* __restrict__ vws,
                       unsigned short* __restrict__ attnws,
                       const int* __restrict__ causal_flag) {
  __shared__ unsigned short Ks[64 * 64];     // [key][d]
  __shared__ unsigned short Vs[64 * 64];     // [d][key] (transposed)
  __shared__ unsigned short Ps[8][16 * 64];  // per-wave P tile

  const int tid  = threadIdx.x;
  const int lane = tid & 31;
  const int wave = tid >> 5;
  const int hf   = lane >> 4;
  const int nlo  = lane & 15;

  const int qblk   = blockIdx.x;
  const int h      = blockIdx.y;
  const int bb     = blockIdx.z;
  const int kvh    = h >> 2;  // GQA group of 4
  const int causal = *causal_flag;

  const int qbase = qblk * 128;
  const int qrow0 = qbase + wave * 16;

  // Preload this wave's Q A-fragments (K-dim = 64 -> two 32-deep steps).
  const unsigned short* qp =
      qws + ((size_t)(bb * SDIM) + qrow0 + nlo) * 2048 + h * 64;
  v16bf aq0 = make_frag(qp + hf * 8, qp + 16 + hf * 8);
  v16bf aq1 = make_frag(qp + 32 + hf * 8, qp + 48 + hf * 8);

  v8f oacc[4] = {};
  float rm[8], rl[8];
#pragma unroll
  for (int r = 0; r < 8; ++r) { rm[r] = -3.0e38f; rl[r] = 0.0f; }

  const int nchunk  = causal ? (qbase / 64 + 2) : (SDIM / 64);
  const float scale = 0.125f;  // 1/sqrt(64)

  for (int kt = 0; kt < nchunk; ++kt) {
    const int key0 = kt * 64;

#if __has_builtin(__builtin_amdgcn_tensor_load_to_lds)
    // --- TDM stage of K chunk: 64 rows x 128B, row stride 1024B -> LDS ---
    if (wave == 0) {
      unsigned long long ga = (unsigned long long)(uintptr_t)(
          kws + ((size_t)(bb * SDIM) + key0) * 512 + kvh * 64);
      u32x4 g0;
      g0[0] = 1u;                                 // count=1, user descriptor
      g0[1] = (unsigned int)(uintptr_t)(&Ks[0]);  // lds_addr (low 32 bits)
      g0[2] = (unsigned int)(ga & 0xFFFFFFFFu);   // global_addr[31:0]
      g0[3] = (unsigned int)((ga >> 32) & 0x1FFFFFFu) | (2u << 30);  // type=2
      i32x8 g1;
      g1[0] = (int)(1u << 16);      // data_size = 2 bytes
      g1[1] = (int)(512u << 16);    // tensor_dim0[15:0] (row length 512 elts)
      g1[2] = (int)(4096u << 16);   // tensor_dim1[15:0] (4096 rows)
      g1[3] = (int)(64u << 16);     // tile_dim0 = 64 (d)
      g1[4] = (int)64u;             // tile_dim1 = 64 (keys)
      g1[5] = (int)512u;            // tensor_dim0_stride = 512 elements
      g1[6] = 0;
      g1[7] = 0;
#if defined(__clang_major__) && __clang_major__ >= 23
      __builtin_amdgcn_tensor_load_to_lds(g0, g1, (i32x4)0, (i32x4)0,
                                          (i32x8)0, 0);
#else
      __builtin_amdgcn_tensor_load_to_lds(g0, g1, (i32x4)0, (i32x4)0, 0);
#endif
      __builtin_amdgcn_s_wait_tensorcnt(0);
    }
#else
    // --- Fallback: cooperative K stage (uint4 loads, b128 LDS stores) ---
#pragma unroll
    for (int i = 0; i < 2; ++i) {
      int e   = i * 256 + tid;  // 512 uint4
      int key = e >> 3;
      int dq  = e & 7;
      uint4 kk = *reinterpret_cast<const uint4*>(
          kws + ((size_t)(bb * SDIM) + key0 + key) * 512 + kvh * 64 + dq * 8);
      *reinterpret_cast<uint4*>(&Ks[key * 64 + dq * 8]) = kk;
    }
#endif
    // --- V stage with transpose into [d][key] ---
#pragma unroll
    for (int i = 0; i < 2; ++i) {
      int e   = i * 256 + tid;  // 512 uint4
      int key = e >> 3;
      int dq  = e & 7;
      uint4 vv = *reinterpret_cast<const uint4*>(
          vws + ((size_t)(bb * SDIM) + key0 + key) * 512 + kvh * 64 + dq * 8);
      Vs[(dq * 8 + 0) * 64 + key] = (unsigned short)(vv.x & 0xFFFFu);
      Vs[(dq * 8 + 1) * 64 + key] = (unsigned short)(vv.x >> 16);
      Vs[(dq * 8 + 2) * 64 + key] = (unsigned short)(vv.y & 0xFFFFu);
      Vs[(dq * 8 + 3) * 64 + key] = (unsigned short)(vv.y >> 16);
      Vs[(dq * 8 + 4) * 64 + key] = (unsigned short)(vv.z & 0xFFFFu);
      Vs[(dq * 8 + 5) * 64 + key] = (unsigned short)(vv.z >> 16);
      Vs[(dq * 8 + 6) * 64 + key] = (unsigned short)(vv.w & 0xFFFFu);
      Vs[(dq * 8 + 7) * 64 + key] = (unsigned short)(vv.w >> 16);
    }
    __syncthreads();

    // S = Q K^T (per-wave 16x64 score slab)
    v8f sacc[4] = {};
#pragma unroll
    for (int t = 0; t < 4; ++t) {
      v16bf bk0 = load_bfrag(&Ks[(16 * t) * 64 + 0], 64, lane);
      sacc[t]   = __builtin_amdgcn_wmma_f32_16x16x32_bf16(
          false, aq0, false, bk0, (short)0, sacc[t], false, false);
      v16bf bk1 = load_bfrag(&Ks[(16 * t) * 64 + 32], 64, lane);
      sacc[t]   = __builtin_amdgcn_wmma_f32_16x16x32_bf16(
          false, aq1, false, bk1, (short)0, sacc[t], false, false);
    }

    // scale + causal mask
#pragma unroll
    for (int t = 0; t < 4; ++t) {
      int keyg = key0 + 16 * t + nlo;
#pragma unroll
      for (int r = 0; r < 8; ++r) {
        float v = sacc[t][r] * scale;
        int qg  = qrow0 + r + hf * 8;
        if (causal && keyg > qg) v = -3.0e38f;
        sacc[t][r] = v;
      }
    }

    // Online softmax (row's 16 N values live across one 16-lane half).
#pragma unroll
    for (int r = 0; r < 8; ++r) {
      float mx = fmaxf(fmaxf(sacc[0][r], sacc[1][r]),
                       fmaxf(sacc[2][r], sacc[3][r]));
#pragma unroll
      for (int off = 1; off < 16; off <<= 1)
        mx = fmaxf(mx, __shfl_xor(mx, off, 32));
      float newm = fmaxf(rm[r], mx);
      float fs   = __expf(rm[r] - newm);
      float sum  = 0.0f;
#pragma unroll
      for (int t = 0; t < 4; ++t) {
        float p    = __expf(sacc[t][r] - newm);
        sacc[t][r] = p;
        sum += p;
      }
#pragma unroll
      for (int off = 1; off < 16; off <<= 1) sum += __shfl_xor(sum, off, 32);
      rl[r] = rl[r] * fs + sum;
      rm[r] = newm;
#pragma unroll
      for (int t = 0; t < 4; ++t) oacc[t][r] *= fs;
    }

    // P -> LDS (bf16, per-wave slab), then O += P V.
#pragma unroll
    for (int t = 0; t < 4; ++t)
#pragma unroll
      for (int r = 0; r < 8; ++r)
        Ps[wave][(r + hf * 8) * 64 + 16 * t + nlo] = f2bf(sacc[t][r]);
    __syncthreads();

    {
      const unsigned short* pb = &Ps[wave][0];
      v16bf ap0 = load_afrag(pb, 64, lane);
      v16bf ap1 = load_afrag(pb + 32, 64, lane);
#pragma unroll
      for (int t = 0; t < 4; ++t) {
        v16bf bv0 = load_bfrag(&Vs[(16 * t) * 64 + 0], 64, lane);
        oacc[t]   = __builtin_amdgcn_wmma_f32_16x16x32_bf16(
            false, ap0, false, bv0, (short)0, oacc[t], false, false);
        v16bf bv1 = load_bfrag(&Vs[(16 * t) * 64 + 32], 64, lane);
        oacc[t]   = __builtin_amdgcn_wmma_f32_16x16x32_bf16(
            false, ap1, false, bv1, (short)0, oacc[t], false, false);
      }
    }
    __syncthreads();
  }

  // Normalize and emit bf16 attn rows [B*S, 2048] for the O projection.
#pragma unroll
  for (int r = 0; r < 8; ++r) {
    float inv = 1.0f / rl[r];
    int row   = bb * SDIM + qrow0 + r + hf * 8;
#pragma unroll
    for (int t = 0; t < 4; ++t)
      attnws[(size_t)row * 2048 + h * 64 + 16 * t + nlo] =
          f2bf(oacc[t][r] * inv);
  }
}

// ---------------------------------------------------------------------------
// Kernel C: attn(bf16)[4096,2048] @ Wo(fp32)[2048,2048] -> fp32 out.
// ---------------------------------------------------------------------------
__global__ __launch_bounds__(256)
void outproj_kernel(const unsigned short* __restrict__ Abf,
                    const float* __restrict__ W, float* __restrict__ out) {
  __shared__ unsigned short As[128 * 32];
  __shared__ unsigned short Bs[64 * 32];

  const int tid  = threadIdx.x;
  const int lane = tid & 31;
  const int wave = tid >> 5;
  const int hf   = lane >> 4;
  const int nlo  = lane & 15;
  const int row0 = blockIdx.x * 128;
  const int col0 = blockIdx.y * 64;

  v8f acc[4] = {};

  for (int kb = 0; kb < HDIM / 32; ++kb) {
#pragma unroll
    for (int i = 0; i < 2; ++i) {  // 512 uint4 of bf16 A tile
      int e  = i * 256 + tid;
      int m  = e >> 2, ku = e & 3;
      uint4 a4 = *reinterpret_cast<const uint4*>(
          Abf + (size_t)(row0 + m) * 2048 + kb * 32 + ku * 8);
      *reinterpret_cast<uint4*>(&As[m * 32 + ku * 8]) = a4;
    }
#pragma unroll
    for (int i = 0; i < 2; ++i) {
      int e = i * 256 + tid;
      int k = e >> 4, n4 = e & 15;
      float4 wv = *reinterpret_cast<const float4*>(
          &W[(size_t)(kb * 32 + k) * 2048 + col0 + n4 * 4]);
      Bs[(n4 * 4 + 0) * 32 + k] = f2bf(wv.x);
      Bs[(n4 * 4 + 1) * 32 + k] = f2bf(wv.y);
      Bs[(n4 * 4 + 2) * 32 + k] = f2bf(wv.z);
      Bs[(n4 * 4 + 3) * 32 + k] = f2bf(wv.w);
    }
    __syncthreads();

    v16bf a = load_afrag(&As[wave * 16 * 32], 32, lane);
#pragma unroll
    for (int t = 0; t < 4; ++t) {
      v16bf b = load_bfrag(&Bs[t * 16 * 32], 32, lane);
      acc[t]  = __builtin_amdgcn_wmma_f32_16x16x32_bf16(
          false, a, false, b, (short)0, acc[t], false, false);
    }
    __syncthreads();
  }

#pragma unroll
  for (int r = 0; r < 8; ++r) {
    int row = row0 + wave * 16 + r + hf * 8;
#pragma unroll
    for (int t = 0; t < 4; ++t)
      out[(size_t)row * 2048 + col0 + nlo + 16 * t] = acc[t][r];
  }
}

// ---------------------------------------------------------------------------
extern "C" void kernel_launch(void* const* d_in, const int* in_sizes, int n_in,
                              void* d_out, int out_size, void* d_ws,
                              size_t ws_size, hipStream_t stream) {
  const float* hidden = (const float*)d_in[0];  // [B,S,H]
  const float* cosT   = (const float*)d_in[1];  // [S,64]
  const float* sinT   = (const float*)d_in[2];
  const float* Wq     = (const float*)d_in[3];  // [2048,2048]
  const float* Wk     = (const float*)d_in[4];  // [2048,512]
  const float* Wv     = (const float*)d_in[5];
  const float* Wo     = (const float*)d_in[6];  // [2048,2048]
  const int* causal   = (const int*)d_in[7];

  float* out      = (float*)d_out;
  float* attn_out = out;                        // 2*2048*2048
  float* k_out    = out + (size_t)4096 * 2048;  // 2*8*2048*64
  float* v_out    = k_out + (size_t)2 * 8 * 2048 * 64;

  unsigned short* ws  = (unsigned short*)d_ws;
  unsigned short* qws = ws;                         // [4096,2048] bf16
  unsigned short* kws = qws + (size_t)4096 * 2048;  // [4096,512]
  unsigned short* vws = kws + (size_t)4096 * 512;   // [4096,512]
  unsigned short* aws = vws + (size_t)4096 * 512;   // [4096,2048]

  dim3 blk(256);
  // Q projection + RoPE (32 head tiles x 32 row tiles)
  proj_rope_kernel<<<dim3(32, 32), blk, 0, stream>>>(hidden, Wq, cosT, sinT,
                                                     2048, 1, qws, nullptr, 32);
  // K projection + RoPE, also emits fp32 k cache [B,NKV,S,64]
  proj_rope_kernel<<<dim3(32, 8), blk, 0, stream>>>(hidden, Wk, cosT, sinT, 512,
                                                    1, kws, k_out, 8);
  // V projection (no RoPE), emits fp32 v cache
  proj_rope_kernel<<<dim3(32, 8), blk, 0, stream>>>(hidden, Wv, cosT, sinT, 512,
                                                    0, vws, v_out, 8);
  // Fused causal GQA flash attention
  flash_attn_kernel<<<dim3(16, 32, 2), blk, 0, stream>>>(qws, kws, vws, aws,
                                                         causal);
  // Output projection
  outproj_kernel<<<dim3(32, 32), blk, 0, stream>>>(aws, Wo, attn_out);
}
